// EncoderDecoder_10814727651472
// MI455X (gfx1250) — matrix operand order repaired and checked by reading, not compiled, not run on hardware
//
#include <hip/hip_runtime.h>
#include <hip/hip_bf16.h>
#include <math.h>

// Problem constants (match reference)
#define SEQ   64
#define BATCH 256
#define VIN   128
#define VOUT  128
#define UH    1024
#define EMB   512
#define TDEC  25

typedef __attribute__((ext_vector_type(16))) _Float16 v16h;
typedef __attribute__((ext_vector_type(8)))  _Float16 v8h;
typedef __attribute__((ext_vector_type(8)))  float    v8f;

// ---------------------------------------------------------------------------
// WMMA GEMM:  C[M,N] = A[M,K] (f16, row-major, lda) * W[N,K]^T (f16, row-major,
// ldw) + bias[N] (f32, optional).  K % 32 == 0, M % 16 == 0, N % 64 == 0.
// One wave computes a 16(M) x 64(N) tile: 1 A-fragment + 4 B-fragments per
// 32-wide K chunk, 4 f32 accumulators, v_wmma_f32_16x16x32_f16.
// ---------------------------------------------------------------------------
__global__ void wmma_gemm_nt(const _Float16* __restrict__ A, int lda,
                             const _Float16* __restrict__ W, int ldw,
                             const float* __restrict__ bias,
                             float* __restrict__ C, int ldc,
                             int M, int N, int K)
{
    const int wid  = (blockIdx.x * blockDim.x + threadIdx.x) >> 5;
    const int lane = threadIdx.x & 31;
    const int tilesN = N >> 6;                 // 64-wide N tiles
    const int total  = (M >> 4) * tilesN;
    if (wid >= total) return;
    const int mbase = (wid / tilesN) << 4;
    const int nbase = (wid % tilesN) << 6;
    const int r  = lane & 15;                  // A row / B(N) column within tile
    const int hi = lane >> 4;                  // half-select per ISA layout

    v8f acc0 = {}, acc1 = {}, acc2 = {}, acc3 = {};

    // A 16x32 f16 layout: hi=0 lane holds K {kb..kb+15};
    //                     hi=1 lane holds K {kb+16..kb+31}.
    const _Float16* Arow = A + (size_t)(mbase + r) * lda + hi * 16;
    // B 32x16 f16 layout: lane(0-15)=n holds K {kb..kb+15} of W row n,
    //                     lane(16-31)   holds K {kb+16..kb+31}.
    const _Float16* Wrow = W + (size_t)(nbase + r) * ldw + hi * 16;
    const size_t wstep = (size_t)16 * ldw;     // next 16-column N sub-tile

    for (int kb = 0; kb < K; kb += 32) {
        v16h a;
        *((v8h*)&a)     = *(const v8h*)(Arow + kb);
        *((v8h*)&a + 1) = *(const v8h*)(Arow + kb + 8);

        const _Float16* wp = Wrow + kb;
        v16h b0, b1, b2, b3;
        *((v8h*)&b0)     = *(const v8h*)(wp);
        *((v8h*)&b0 + 1) = *(const v8h*)(wp + 8);
        wp += wstep;
        *((v8h*)&b1)     = *(const v8h*)(wp);
        *((v8h*)&b1 + 1) = *(const v8h*)(wp + 8);
        wp += wstep;
        *((v8h*)&b2)     = *(const v8h*)(wp);
        *((v8h*)&b2 + 1) = *(const v8h*)(wp + 8);
        wp += wstep;
        *((v8h*)&b3)     = *(const v8h*)(wp);
        *((v8h*)&b3 + 1) = *(const v8h*)(wp + 8);

        acc0 = __builtin_amdgcn_wmma_f32_16x16x32_f16(false, a, false, b0,
                                                      (short)0, acc0, false, false);
        acc1 = __builtin_amdgcn_wmma_f32_16x16x32_f16(false, a, false, b1,
                                                      (short)0, acc1, false, false);
        acc2 = __builtin_amdgcn_wmma_f32_16x16x32_f16(false, a, false, b2,
                                                      (short)0, acc2, false, false);
        acc3 = __builtin_amdgcn_wmma_f32_16x16x32_f16(false, a, false, b3,
                                                      (short)0, acc3, false, false);
    }

    // C/D layout: acc element e -> row mbase + e + 8*hi, col nbase + 16*j + r.
    v8f accs[4] = {acc0, acc1, acc2, acc3};
#pragma unroll
    for (int j = 0; j < 4; ++j) {
        const int col = nbase + j * 16 + r;
        const float bv = bias ? bias[col] : 0.f;
#pragma unroll
        for (int e = 0; e < 8; ++e) {
            const int row = mbase + e + 8 * hi;
            C[(size_t)row * ldc + col] = accs[j][e] + bv;
        }
    }
}

// ---------------------------------------------------------------------------
// Elementwise / reduction helpers
// ---------------------------------------------------------------------------
__global__ void cvt_f32_f16(const float* __restrict__ src,
                            _Float16* __restrict__ dst, int n)
{
    int i = blockIdx.x * blockDim.x + threadIdx.x;
    if (i < n) dst[i] = (_Float16)src[i];
}

__global__ void init_state(float* __restrict__ h, _Float16* __restrict__ h_h,
                           float* __restrict__ out0, int* __restrict__ tok)
{
    int i = blockIdx.x * blockDim.x + threadIdx.x;
    if (i < BATCH * UH)   { h[i] = 0.f; h_h[i] = (_Float16)0.f; }
    if (i < BATCH * VOUT) { out0[i] = ((i % VOUT) == 0) ? 1.f : 0.f; }
    if (i < BATCH)        { tok[i] = 0; }
}

__device__ __forceinline__ float sigmoidf(float x) { return 1.f / (1.f + expf(-x)); }

// Encoder GRU gate combine + length mask + masked output write.
__global__ void gru_combine_enc(const float* __restrict__ gi,
                                const float* __restrict__ gh,
                                float* __restrict__ h,
                                _Float16* __restrict__ h_h,
                                _Float16* __restrict__ enc_out_h,
                                const int* __restrict__ x_len, int t)
{
    int i = blockIdx.x * blockDim.x + threadIdx.x;
    if (i >= BATCH * UH) return;
    int b = i / UH, u = i - b * UH;
    const float* gib = gi + (size_t)b * 3 * UH;
    const float* ghb = gh + (size_t)b * 3 * UH;
    float r = sigmoidf(gib[u] + ghb[u]);
    float z = sigmoidf(gib[UH + u] + ghb[UH + u]);
    float n = tanhf(gib[2 * UH + u] + r * ghb[2 * UH + u]);
    float hold = h[i];
    float hn = (1.f - z) * n + z * hold;
    bool m = (t < x_len[b]);
    float hnew = m ? hn : hold;
    h[i] = hnew;
    h_h[i] = (_Float16)hnew;
    enc_out_h[((size_t)b * SEQ + t) * UH + u] = (_Float16)(m ? hn : 0.f);
}

// Decoder GRU gate combine (no mask).
__global__ void gru_combine_dec(const float* __restrict__ gi,
                                const float* __restrict__ gh,
                                float* __restrict__ h,
                                _Float16* __restrict__ h_h)
{
    int i = blockIdx.x * blockDim.x + threadIdx.x;
    if (i >= BATCH * UH) return;
    int b = i / UH, u = i - b * UH;
    const float* gib = gi + (size_t)b * 3 * UH;
    const float* ghb = gh + (size_t)b * 3 * UH;
    float r = sigmoidf(gib[u] + ghb[u]);
    float z = sigmoidf(gib[UH + u] + ghb[UH + u]);
    float n = tanhf(gib[2 * UH + u] + r * ghb[2 * UH + u]);
    float hn = (1.f - z) * n + z * h[i];
    h[i] = hn;
    h_h[i] = (_Float16)hn;
}

// logits[b,s] = V . tanh(enc_proj[b,s,:] + w2h[b,:]) + Vb  (one wave per (b,s))
__global__ void attn_logits(const float* __restrict__ enc_proj,
                            const float* __restrict__ w2h,
                            const float* __restrict__ Vw,
                            const float* __restrict__ Vb,
                            float* __restrict__ logits)
{
    int wid  = (blockIdx.x * blockDim.x + threadIdx.x) >> 5;
    int lane = threadIdx.x & 31;
    if (wid >= BATCH * SEQ) return;
    int b = wid / SEQ, s = wid - b * SEQ;
    const float* ep = enc_proj + ((size_t)b * SEQ + s) * UH;
    const float* wh = w2h + (size_t)b * UH;
    float acc = 0.f;
    for (int u = lane; u < UH; u += 32)
        acc += tanhf(ep[u] + wh[u]) * Vw[u];
#pragma unroll
    for (int off = 16; off > 0; off >>= 1)
        acc += __shfl_down(acc, off, 32);
    if (lane == 0) logits[wid] = acc + Vb[0];
}

// softmax over S per batch row (block = 64 threads).
__global__ void attn_softmax(const float* __restrict__ logits,
                             float* __restrict__ attn)
{
    __shared__ float sm[SEQ];
    int b = blockIdx.x, s = threadIdx.x;
    float v = logits[b * SEQ + s];
    sm[s] = v; __syncthreads();
    for (int off = 32; off > 0; off >>= 1) {
        if (s < off) sm[s] = fmaxf(sm[s], sm[s + off]);
        __syncthreads();
    }
    float mx = sm[0]; __syncthreads();
    float e = expf(v - mx);
    sm[s] = e; __syncthreads();
    for (int off = 32; off > 0; off >>= 1) {
        if (s < off) sm[s] += sm[s + off];
        __syncthreads();
    }
    attn[b * SEQ + s] = e / sm[0];
}

// ctx[b,u] = sum_s attn[b,s]*enc_out[b,s,u]; written f16 into gx[:, 0:UH].
__global__ void attn_ctx(const float* __restrict__ attn,
                         const _Float16* __restrict__ enc_out_h,
                         _Float16* __restrict__ gx)
{
    int i = blockIdx.x * blockDim.x + threadIdx.x;
    if (i >= BATCH * UH) return;
    int b = i / UH, u = i - b * UH;
    const _Float16* eo = enc_out_h + (size_t)b * SEQ * UH + u;
    const float* ab = attn + b * SEQ;
    float acc = 0.f;
#pragma unroll 8
    for (int s = 0; s < SEQ; ++s)
        acc += ab[s] * (float)eo[(size_t)s * UH];
    gx[(size_t)b * (UH + EMB) + u] = (_Float16)acc;
}

// emb = onehot(tok) @ o2h_W^T + o2h_b  ==  o2h_W[:, tok] + o2h_b (column gather)
__global__ void embed_tok(const int* __restrict__ tok,
                          const float* __restrict__ o2hW,
                          const float* __restrict__ o2hb,
                          _Float16* __restrict__ gx)
{
    int i = blockIdx.x * blockDim.x + threadIdx.x;
    if (i >= BATCH * EMB) return;
    int b = i / EMB, e = i - b * EMB;
    float v = o2hW[(size_t)e * VOUT + tok[b]] + o2hb[e];
    gx[(size_t)b * (UH + EMB) + UH + e] = (_Float16)v;
}

// argmax over VOUT=128 (first index wins on ties, matching jnp.argmax)
__global__ void argmax_k(const float* __restrict__ pred, int* __restrict__ tok)
{
    __shared__ float sv[VOUT];
    __shared__ int   si[VOUT];
    int b = blockIdx.x, v = threadIdx.x;
    sv[v] = pred[(size_t)b * VOUT + v];
    si[v] = v;
    __syncthreads();
    for (int off = VOUT / 2; off > 0; off >>= 1) {
        if (v < off) {
            if (sv[v + off] > sv[v] ||
                (sv[v + off] == sv[v] && si[v + off] < si[v])) {
                sv[v] = sv[v + off];
                si[v] = si[v + off];
            }
        }
        __syncthreads();
    }
    if (v == 0) tok[b] = si[0];
}

// ---------------------------------------------------------------------------
// Host side
// ---------------------------------------------------------------------------
static inline size_t alignUp(size_t x) { return (x + 255) & ~(size_t)255; }

static inline void launch_gemm(hipStream_t s, const _Float16* A, int lda,
                               const _Float16* W, int ldw, const float* bias,
                               float* C, int ldc, int M, int N, int K)
{
    int waves = (M / 16) * (N / 64);
    int threads = waves * 32;
    int grid = (threads + 255) / 256;
    hipLaunchKernelGGL(wmma_gemm_nt, dim3(grid), dim3(256), 0, s,
                       A, lda, W, ldw, bias, C, ldc, M, N, K);
}

static inline void launch_cvt(hipStream_t s, const float* src, _Float16* dst, int n)
{
    hipLaunchKernelGGL(cvt_f32_f16, dim3((n + 255) / 256), dim3(256), 0, s,
                       src, dst, n);
}

extern "C" void kernel_launch(void* const* d_in, const int* in_sizes, int n_in,
                              void* d_out, int out_size, void* d_ws, size_t ws_size,
                              hipStream_t stream)
{
    (void)in_sizes; (void)n_in; (void)out_size; (void)ws_size;

    // Inputs per setup_inputs() order
    const float* x        = (const float*)d_in[0];   // [S,B,VIN]
    const int*   x_len    = (const int*)  d_in[1];   // [B]
    const float* enc_Wih  = (const float*)d_in[2];   // [3U,VIN]
    const float* enc_Whh  = (const float*)d_in[3];   // [3U,U]
    const float* enc_bih  = (const float*)d_in[4];
    const float* enc_bhh  = (const float*)d_in[5];
    const float* dec_Wih  = (const float*)d_in[6];   // [3U,E+U]
    const float* dec_Whh  = (const float*)d_in[7];   // [3U,U]
    const float* dec_bih  = (const float*)d_in[8];
    const float* dec_bhh  = (const float*)d_in[9];
    const float* o2h_W    = (const float*)d_in[10];  // [E,VOUT]
    const float* o2h_b    = (const float*)d_in[11];
    const float* fc_W     = (const float*)d_in[12];  // [VOUT,U]
    const float* fc_b     = (const float*)d_in[13];
    const float* W1_W     = (const float*)d_in[14];  // [U,U]
    const float* W1_b     = (const float*)d_in[15];
    const float* W2_W     = (const float*)d_in[16];  // [U,U]
    const float* W2_b     = (const float*)d_in[17];
    const float* V_W      = (const float*)d_in[18];  // [1,U]
    const float* V_b      = (const float*)d_in[19];  // [1]

    float* out = (float*)d_out;                      // [25,B,VOUT] f32

    // ---- workspace carving ----
    char* p = (char*)d_ws;
    auto takeH = [&](size_t nelem) { _Float16* q = (_Float16*)p; p += alignUp(nelem * 2); return q; };
    auto takeF = [&](size_t nelem) { float*    q = (float*)p;    p += alignUp(nelem * 4); return q; };
    auto takeI = [&](size_t nelem) { int*      q = (int*)p;      p += alignUp(nelem * 4); return q; };

    _Float16* x_h       = takeH((size_t)SEQ * BATCH * VIN);
    _Float16* encWih_h  = takeH((size_t)3 * UH * VIN);
    _Float16* encWhh_h  = takeH((size_t)3 * UH * UH);
    _Float16* decWih_h  = takeH((size_t)3 * UH * (UH + EMB));
    _Float16* decWhh_h  = takeH((size_t)3 * UH * UH);
    _Float16* fc_h      = takeH((size_t)VOUT * UH);
    _Float16* W1_h      = takeH((size_t)UH * UH);
    _Float16* W2_h      = takeH((size_t)UH * UH);
    float*    gi        = takeF((size_t)BATCH * 3 * UH);
    float*    gh        = takeF((size_t)BATCH * 3 * UH);
    float*    h_f32     = takeF((size_t)BATCH * UH);
    _Float16* h_h       = takeH((size_t)BATCH * UH);
    _Float16* enc_out_h = takeH((size_t)BATCH * SEQ * UH);
    float*    enc_proj  = takeF((size_t)BATCH * SEQ * UH);
    float*    w2h       = takeF((size_t)BATCH * UH);
    float*    logits    = takeF((size_t)BATCH * SEQ);
    float*    attn      = takeF((size_t)BATCH * SEQ);
    _Float16* gx        = takeH((size_t)BATCH * (UH + EMB));
    int*      tok       = takeI(BATCH);

    // ---- one-time f32 -> f16 conversions (deterministic each call) ----
    launch_cvt(stream, x,       x_h,      SEQ * BATCH * VIN);
    launch_cvt(stream, enc_Wih, encWih_h, 3 * UH * VIN);
    launch_cvt(stream, enc_Whh, encWhh_h, 3 * UH * UH);
    launch_cvt(stream, dec_Wih, decWih_h, 3 * UH * (UH + EMB));
    launch_cvt(stream, dec_Whh, decWhh_h, 3 * UH * UH);
    launch_cvt(stream, fc_W,    fc_h,     VOUT * UH);
    launch_cvt(stream, W1_W,    W1_h,     UH * UH);
    launch_cvt(stream, W2_W,    W2_h,     UH * UH);

    // h0 = 0, out[0] = start one-hot, tok = START_CODE(0)
    hipLaunchKernelGGL(init_state, dim3((BATCH * UH + 255) / 256), dim3(256), 0,
                       stream, h_f32, h_h, out, tok);

    const int euB = (BATCH * UH + 255) / 256;

    // ---- encoder: 64 masked GRU steps ----
    for (int t = 0; t < SEQ; ++t) {
        // gi = x_t @ enc_Wih^T + bih   [256,3072,128]
        launch_gemm(stream, x_h + (size_t)t * BATCH * VIN, VIN,
                    encWih_h, VIN, enc_bih, gi, 3 * UH, BATCH, 3 * UH, VIN);
        // gh = h @ enc_Whh^T + bhh     [256,3072,1024]
        launch_gemm(stream, h_h, UH, encWhh_h, UH, enc_bhh, gh, 3 * UH,
                    BATCH, 3 * UH, UH);
        hipLaunchKernelGGL(gru_combine_enc, dim3(euB), dim3(256), 0, stream,
                           gi, gh, h_f32, h_h, enc_out_h, x_len, t);
    }

    // enc_proj = enc_out @ W1^T + W1_b   [16384,1024,1024]
    launch_gemm(stream, enc_out_h, UH, W1_h, UH, W1_b, enc_proj, UH,
                BATCH * SEQ, UH, UH);

    // ---- greedy decoder: 24 steps ----
    for (int t = 0; t < TDEC - 1; ++t) {
        // w2h = h @ W2^T + W2_b
        launch_gemm(stream, h_h, UH, W2_h, UH, W2_b, w2h, UH, BATCH, UH, UH);
        // attention
        hipLaunchKernelGGL(attn_logits,
                           dim3((BATCH * SEQ * 32 + 255) / 256), dim3(256), 0,
                           stream, enc_proj, w2h, V_W, V_b, logits);
        hipLaunchKernelGGL(attn_softmax, dim3(BATCH), dim3(SEQ), 0, stream,
                           logits, attn);
        hipLaunchKernelGGL(attn_ctx, dim3(euB), dim3(256), 0, stream,
                           attn, enc_out_h, gx);
        // emb (one-hot matmul == column gather)
        hipLaunchKernelGGL(embed_tok, dim3((BATCH * EMB + 255) / 256), dim3(256),
                           0, stream, tok, o2h_W, o2h_b, gx);
        // gi = gx @ dec_Wih^T + bih  [256,3072,1536]
        launch_gemm(stream, gx, UH + EMB, decWih_h, UH + EMB, dec_bih, gi,
                    3 * UH, BATCH, 3 * UH, UH + EMB);
        // gh = h @ dec_Whh^T + bhh   [256,3072,1024]
        launch_gemm(stream, h_h, UH, decWhh_h, UH, dec_bhh, gh, 3 * UH,
                    BATCH, 3 * UH, UH);
        hipLaunchKernelGGL(gru_combine_dec, dim3(euB), dim3(256), 0, stream,
                           gi, gh, h_f32, h_h);
        // pred = h @ fc_W^T + fc_b -> out[t+1]   [256,128,1024]
        launch_gemm(stream, h_h, UH, fc_h, UH, fc_b,
                    out + (size_t)(t + 1) * BATCH * VOUT, VOUT,
                    BATCH, VOUT, UH);
        // feedback token
        hipLaunchKernelGGL(argmax_k, dim3(BATCH), dim3(VOUT), 0, stream,
                           out + (size_t)(t + 1) * BATCH * VOUT, tok);
    }
}
